// GConv_12618613915757
// MI455X (gfx1250) — matrix operand order repaired and checked by reading, not compiled, not run on hardware
//
#include <hip/hip_runtime.h>
#include <hip/hip_bf16.h>

typedef __attribute__((ext_vector_type(16))) _Float16 v16h;
typedef __attribute__((ext_vector_type(8)))  float    v8f;

#define DIM 256
#define CHUNK_HALFS 8192          // 16 n-tiles * 32 lanes * 16 halfs (hi or lo)
#define BUF_HALFS   (2 * CHUNK_HALFS)   // hi + lo for one chunk = 32 KB

// ---------------------------------------------------------------------------
// Zero scratch accumulators (sum/cnt buffers) — capture-safe.
// ---------------------------------------------------------------------------
__global__ __launch_bounds__(256) void zero_f32(float* __restrict__ p, long n) {
    long i = (long)blockIdx.x * blockDim.x + threadIdx.x;
    long stride = (long)gridDim.x * blockDim.x;
    for (; i < n; i += stride) p[i] = 0.0f;
}

// ---------------------------------------------------------------------------
// Edge scatter: sum[dst[e]][:] += x[src[e]][:] ; cnt[dst[e]] += 1
// One 256-thread block per edge; coalesced 1KB row read; fp32 atomics land in
// the 192MB L2 (target region is 40MB / 8MB — fully L2 resident).
// ---------------------------------------------------------------------------
__global__ __launch_bounds__(256) void sage_scatter(
    const float* __restrict__ xsrc, const int* __restrict__ src,
    const int* __restrict__ dst, float* __restrict__ sum,
    float* __restrict__ cnt, int E)
{
    int e = blockIdx.x;
    if (e >= E) return;
    int s = src[e];
    int d = dst[e];
    int t = threadIdx.x;
    atomicAdd(&sum[(size_t)d * DIM + t], xsrc[(size_t)s * DIM + t]);
    if (t == 0) atomicAdd(&cnt[d], 1.0f);
}

// ---------------------------------------------------------------------------
// Weight pre-pack: split fp32 W (KxN, 256x256) into hi/lo f16 stored in the
// wave32 WMMA B-fragment order used here (contiguous k = 16h + j, same as the A loads):
//   element j of frag (chunk, n, lane) <-> W[chunk*32 + 16*(lane>>4) + j][n*16 + (lane&15)]
//   linear half index = ((chunk*16 + n)*32 + lane)*16 + j
// Chunk c occupies halfs [c*8192, (c+1)*8192): contiguous, LDS-stageable.
// ---------------------------------------------------------------------------
__global__ __launch_bounds__(256) void pack_w_frags(
    const float* __restrict__ W, _Float16* __restrict__ hi,
    _Float16* __restrict__ lo)
{
    int idx   = blockIdx.x * 256 + threadIdx.x;   // 0..65535 (grid 256)
    int j     = idx & 15;
    int lane  = (idx >> 4) & 31;
    int n     = (idx >> 9) & 15;
    int chunk = (idx >> 13) & 7;
    int half  = lane >> 4;
    int lmod  = lane & 15;
    int krow  = chunk * 32 + 16 * half + j;
    int col   = n * 16 + lmod;
    float f   = W[(size_t)krow * DIM + col];
    _Float16 h = (_Float16)f;
    hi[idx] = h;
    lo[idx] = (_Float16)(f - (float)h);
}

// ---------------------------------------------------------------------------
// A fragment (16-bit A 16x32, MxK): lane<16 row=M, elems 0..15 = K kb+16h+0..15
// (h = lane>=16). hi/lo split for fp32-class
// accuracy from three f16 WMMAs (ahi*blo + alo*bhi + ahi*bhi), f32 accum.
// ---------------------------------------------------------------------------
__device__ __forceinline__ void load_a_frag(const float* __restrict__ row,
                                            int kb, int half, float scale,
                                            v16h& hi, v16h& lo)
{
    const float4* p = (const float4*)(row + kb + 16 * half);       // 64B aligned
    float av[16];
    *(float4*)(av + 0)  = p[0];
    *(float4*)(av + 4)  = p[1];
    *(float4*)(av + 8)  = p[2];
    *(float4*)(av + 12) = p[3];
#pragma unroll
    for (int i = 0; i < 16; ++i) {
        float f = av[i] * scale;
        _Float16 h = (_Float16)f;
        hi[i] = h;
        lo[i] = (_Float16)(f - (float)h);
    }
}

// ---------------------------------------------------------------------------
// Fused SAGE layer tail:
//   out = PReLU( (sum/max(cnt,1)) @ Wl + Xr @ Wr + bias, alpha )
// Block = 8 waves, one 16-row M tile per wave, full N=256 in 128 acc VGPRs.
// B fragments cooperatively double-buffer-staged through LDS (32KB/chunk,
// 64KB total): global->regs prefetch of chunk c+1 is issued before the 48
// WMMAs of chunk c (read via ds_load_b128), regs->LDS store lands after,
// one barrier per chunk. Cuts weight L2 traffic 8x and hides load latency.
// ---------------------------------------------------------------------------
__global__ __launch_bounds__(256) void sage_gemm_prelu(
    const float* __restrict__ sumBuf, const float* __restrict__ cnt,
    const float* __restrict__ xr,
    const _Float16* __restrict__ Wlhi, const _Float16* __restrict__ Wllo,
    const _Float16* __restrict__ Wrhi, const _Float16* __restrict__ Wrlo,
    const float* __restrict__ bias, const float* __restrict__ alpha,
    float* __restrict__ out, int M)
{
    extern __shared__ _Float16 smem[];          // 2 * BUF_HALFS halfs = 64 KB

    const int t    = threadIdx.x;
    const int lane = t & 31;
    const int wave = t >> 5;
    const int tile = blockIdx.x * 8 + wave;
    const bool active = (tile * 16 < M);        // wave-uniform

    const int half = lane >> 4;
    const int lmod = lane & 15;
    const int rowA = active ? (tile * 16 + lmod) : lmod;   // safe row

    const float inv = 1.0f / fmaxf(cnt[rowA], 1.0f);
    const float* __restrict__ aggRow = sumBuf + (size_t)rowA * DIM;
    const float* __restrict__ xRow   = xr     + (size_t)rowA * DIM;

    v8f acc[16];
#pragma unroll
    for (int n = 0; n < 16; ++n)
        acc[n] = (v8f){0.f, 0.f, 0.f, 0.f, 0.f, 0.f, 0.f, 0.f};

    // ---- stage phase 0 (pass 0 / chunk 0) ----
    uint4 rh[4], rl[4];
    {
        const uint4* hs = (const uint4*)Wlhi;
        const uint4* ls = (const uint4*)Wllo;
#pragma unroll
        for (int r = 0; r < 4; ++r) { rh[r] = hs[t + r * 256]; rl[r] = ls[t + r * 256]; }
        uint4* dh = (uint4*)smem;
        uint4* dl = (uint4*)(smem + CHUNK_HALFS);
#pragma unroll
        for (int r = 0; r < 4; ++r) { dh[t + r * 256] = rh[r]; dl[t + r * 256] = rl[r]; }
    }
    __syncthreads();

    // ---- main pipeline: 16 phases = 2 passes (W_l then W_r) x 8 K-chunks ----
#pragma unroll 1
    for (int phase = 0; phase < 16; ++phase) {
        const int np = phase + 1;

        // prefetch next chunk's packed fragments into registers
        if (np < 16) {
            const _Float16* hsrc = ((np >> 3) ? Wrhi : Wlhi) + (size_t)(np & 7) * CHUNK_HALFS;
            const _Float16* lsrc = ((np >> 3) ? Wrlo : Wllo) + (size_t)(np & 7) * CHUNK_HALFS;
            const uint4* hs = (const uint4*)hsrc;
            const uint4* ls = (const uint4*)lsrc;
#pragma unroll
            for (int r = 0; r < 4; ++r) { rh[r] = hs[t + r * 256]; rl[r] = ls[t + r * 256]; }
        }

        // compute current chunk from LDS
        if (active) {
            const int pass = phase >> 3;
            const int c    = phase & 7;
            const float* Arow  = pass ? xRow : aggRow;
            const float  scale = pass ? 1.0f : inv;
            v16h ahi, alo;
            load_a_frag(Arow, c * 32, half, scale, ahi, alo);

            const _Float16* base = smem + (size_t)(phase & 1) * BUF_HALFS;
            const v16h* __restrict__ bh = (const v16h*)base + lane;
            const v16h* __restrict__ bl = (const v16h*)(base + CHUNK_HALFS) + lane;
#pragma unroll
            for (int n = 0; n < 16; ++n) {
                v16h bhi = bh[n * 32];
                v16h blo = bl[n * 32];
                acc[n] = __builtin_amdgcn_wmma_f32_16x16x32_f16(
                    false, ahi, false, blo, (short)0, acc[n], false, false);
                acc[n] = __builtin_amdgcn_wmma_f32_16x16x32_f16(
                    false, alo, false, bhi, (short)0, acc[n], false, false);
                acc[n] = __builtin_amdgcn_wmma_f32_16x16x32_f16(
                    false, ahi, false, bhi, (short)0, acc[n], false, false);
            }
        }

        // commit prefetched regs to the other LDS buffer
        if (np < 16) {
            _Float16* dbase = smem + (size_t)(np & 1) * BUF_HALFS;
            uint4* dh = (uint4*)dbase;
            uint4* dl = (uint4*)(dbase + CHUNK_HALFS);
#pragma unroll
            for (int r = 0; r < 4; ++r) { dh[t + r * 256] = rh[r]; dl[t + r * 256] = rl[r]; }
        }
        __syncthreads();
    }

    // ---- epilogue: bias + PReLU + store ----
    // C/D layout: acc[n] vgpr i -> row (tile*16 + half*8 + i), col n*16+lmod.
    if (active) {
#pragma unroll
        for (int n = 0; n < 16; ++n) {
            int col = n * 16 + lmod;
            float bv = bias[col];
            float av = alpha[col];
            float* orow = out + (size_t)(tile * 16 + half * 8) * DIM + col;
#pragma unroll
            for (int i = 0; i < 8; ++i) {
                float v = acc[n][i] + bv;
                v = v > 0.0f ? v : av * v;
                orow[(size_t)i * DIM] = v;
            }
        }
    }
}

// ---------------------------------------------------------------------------
extern "C" void kernel_launch(void* const* d_in, const int* in_sizes, int n_in,
                              void* d_out, int out_size, void* d_ws, size_t ws_size,
                              hipStream_t stream)
{
    const float* x    = (const float*)d_in[0];
    const int*   src1 = (const int*)  d_in[1];
    const int*   dst1 = (const int*)  d_in[2];
    const int*   src2 = (const int*)  d_in[3];
    const int*   dst2 = (const int*)  d_in[4];
    // d_in[5]=n1, d_in[6]=n2: 1-element device ints; sizes fixed by reference
    const float* Wl1 = (const float*)d_in[7];
    const float* Wr1 = (const float*)d_in[8];
    const float* b1  = (const float*)d_in[9];
    const float* a1  = (const float*)d_in[10];
    const float* Wl2 = (const float*)d_in[11];
    const float* Wr2 = (const float*)d_in[12];
    const float* b2  = (const float*)d_in[13];
    const float* a2  = (const float*)d_in[14];

    const int N1 = 40000, N2 = 8000;
    const int E1 = in_sizes[1];
    const int E2 = in_sizes[3];

    float* ws   = (float*)d_ws;
    float* sum1 = ws;                            // N1*256
    float* cnt1 = sum1 + (size_t)N1 * DIM;       // N1
    float* sum2 = cnt1 + N1;                     // N2*256
    float* cnt2 = sum2 + (size_t)N2 * DIM;       // N2
    float* h1   = cnt2 + N2;                     // N1*256 (no zero needed)

    // Packed weight fragments (hi/lo f16), 64K halfs per array; region starts
    // at a 32B-aligned byte offset (22,576,000 floats * 4 = 32 * 2,822,000).
    _Float16* pack = (_Float16*)(h1 + (size_t)N1 * DIM);
    const size_t WSZ = (size_t)DIM * DIM;        // 65536 halfs per array
    _Float16* Wl1hi = pack + 0 * WSZ;
    _Float16* Wl1lo = pack + 1 * WSZ;
    _Float16* Wr1hi = pack + 2 * WSZ;
    _Float16* Wr1lo = pack + 3 * WSZ;
    _Float16* Wl2hi = pack + 4 * WSZ;
    _Float16* Wl2lo = pack + 5 * WSZ;
    _Float16* Wr2hi = pack + 6 * WSZ;
    _Float16* Wr2lo = pack + 7 * WSZ;

    long nzero = (long)N1 * DIM + N1 + (long)N2 * DIM + N2;
    zero_f32<<<2048, 256, 0, stream>>>(ws, nzero);

    pack_w_frags<<<256, 256, 0, stream>>>(Wl1, Wl1hi, Wl1lo);
    pack_w_frags<<<256, 256, 0, stream>>>(Wr1, Wr1hi, Wr1lo);
    pack_w_frags<<<256, 256, 0, stream>>>(Wl2, Wl2hi, Wl2lo);
    pack_w_frags<<<256, 256, 0, stream>>>(Wr2, Wr2hi, Wr2lo);

    const size_t shmem = 2 * (size_t)BUF_HALFS * sizeof(_Float16);   // 64 KB

    // Layer 1
    sage_scatter<<<E1, 256, 0, stream>>>(x, src1, dst1, sum1, cnt1, E1);
    sage_gemm_prelu<<<(N1 / 16 + 7) / 8, 256, shmem, stream>>>(
        sum1, cnt1, x, Wl1hi, Wl1lo, Wr1hi, Wr1lo, b1, a1, h1, N1);

    // Layer 2
    sage_scatter<<<E2, 256, 0, stream>>>(h1, src2, dst2, sum2, cnt2, E2);
    sage_gemm_prelu<<<(N2 / 16 + 7) / 8, 256, shmem, stream>>>(
        sum2, cnt2, h1, Wl2hi, Wl2lo, Wr2hi, Wr2lo, b2, a2, (float*)d_out, N2);
}
